// Transformer_42374147342780
// MI455X (gfx1250) — matrix operand order repaired and checked by reading, not compiled, not run on hardware
//
#include <hip/hip_runtime.h>
#include <hip/hip_bf16.h>
#include <math.h>

// ---------------------------------------------------------------------------
// MI455X (gfx1250) transformer forward, round 3 (fix async builtin arg types).
//  - All matmuls on v_wmma_f32_16x16x32_f16 (f16 operands, f32 accumulate).
//  - Weights converted ONCE to pre-transposed f16 [N][K]; activations kept as
//    f16 copies so every GEMM/attention tile load is a raw byte copy.
//  - Tile staging via GLOBAL_LOAD_ASYNC_TO_LDS (ASYNCcnt) with double-buffered
//    LDS in the GEMM (copy tile i+1 overlaps WMMA on tile i).
//  - Flash attention: no S x S materialization; V stored pre-transposed.
// ---------------------------------------------------------------------------

typedef __attribute__((ext_vector_type(16))) _Float16 v16h;
typedef __attribute__((ext_vector_type(4)))  _Float16 v4h;
typedef __attribute__((ext_vector_type(8)))  float    v8f;
typedef int v2i __attribute__((ext_vector_type(2)));
typedef int v4i __attribute__((ext_vector_type(4)));

#define NL    4
#define NH    8
#define DM    1024
#define DK    128
#define NB    4
#define SEQ   2048
#define MTOK  (NB * SEQ)
#define NEGV  (-1.0e9f)

// ---------------- Async global->LDS copy (CDNA5) ---------------------------

#if __has_builtin(__builtin_amdgcn_global_load_async_to_lds_b64)
#define HAVE_ASYNC_LDS 1
#else
#define HAVE_ASYNC_LDS 0
#endif

typedef __attribute__((address_space(1))) v2i gv2i;   // global src, b64
typedef __attribute__((address_space(3))) v2i lv2i;   // LDS dst,   b64
typedef __attribute__((address_space(1))) v4i gv4i;   // global src, b128
typedef __attribute__((address_space(3))) v4i lv4i;   // LDS dst,   b128

__device__ inline void async_b64(const _Float16* g, _Float16* l) {
#if HAVE_ASYNC_LDS
  __builtin_amdgcn_global_load_async_to_lds_b64((gv2i*)(void*)g,
                                                (lv2i*)(void*)l, 0, 0);
#else
  *(uint2*)l = *(const uint2*)g;
#endif
}
__device__ inline void async_b128(const void* g, void* l) {
#if HAVE_ASYNC_LDS
  __builtin_amdgcn_global_load_async_to_lds_b128((gv4i*)(void*)g,
                                                 (lv4i*)(void*)l, 0, 0);
#else
  *(uint4*)l = *(const uint4*)g;
#endif
}

#if __has_builtin(__builtin_amdgcn_s_wait_asynccnt)
#define WAIT_ASYNC(n) __builtin_amdgcn_s_wait_asynccnt(n)
#else
#define WAIT_ASYNC(n) asm volatile("s_wait_asynccnt %0" ::"i"(n) : "memory")
#endif

// ---------------- WMMA helpers (CDNA5 VGPR layouts) ------------------------

__device__ inline v8f wmma_f16(v16h a, v16h b, v8f c) {
  return __builtin_amdgcn_wmma_f32_16x16x32_f16(false, a, false, b,
                                                (short)0, c, false, false);
}
__device__ inline v8f v8f_zero() {
  v8f z;
#pragma unroll
  for (int e = 0; e < 8; ++e) z[e] = 0.0f;
  return z;
}

// A fragment (16x32, M x K) from row-major LDS. Lane holds row (lane&15);
// lanes 0-15: K 0..15, lanes 16-31: K 16..31 (contiguous).
__device__ inline v16h lds_fragA(const _Float16* m, int row0, int k0, int ld) {
  const int lane = threadIdx.x & 31;
  const _Float16* p = m + (size_t)(row0 + (lane & 15)) * ld + k0 + ((lane >> 4) << 4);
  v16h f;
#pragma unroll
  for (int e = 0; e < 16; ++e) f[e] = p[e];
  return f;
}
// B fragment (32x16, K x N) from LDS stored [N][K]. Lane holds col (lane&15);
// lanes 0-15: K 0..15, lanes 16-31: K 16..31 (contiguous).
__device__ inline v16h lds_fragB(const _Float16* m, int n0, int k0, int ld) {
  const int lane = threadIdx.x & 31;
  const _Float16* p = m + (size_t)(n0 + (lane & 15)) * ld + k0 + ((lane >> 4) << 4);
  v16h f;
#pragma unroll
  for (int e = 0; e < 16; ++e) f[e] = p[e];
  return f;
}

// Copy a 64-row x 32-half tile (row-major, gld element stride) into LDS (ld=40).
__device__ inline void tile_copy64x32(const _Float16* g, size_t gld, _Float16* l) {
  const int tid = threadIdx.x;
#pragma unroll
  for (int it = 0; it < 4; ++it) {
    int idx = tid + it * 128;
    int r = idx >> 3, seg = (idx & 7) << 2;            // 8B segments
    async_b64(g + (size_t)r * gld + seg, l + r * 40 + seg);
  }
}

// ---------------- f16 WMMA GEMM, double-buffered async staging -------------
// C = epi((A[M][K] * Bt^T + bias) * outScale); A f16 row-major, Bt f16 [N][K].
// Outputs: Cf (f32, optional), C16 (f16, optional; outT=1 writes V transposed
// as [B][H][DK][SEQ]). Block 128 thr (4 waves), 64x64 tile.

__global__ __launch_bounds__(128)
void gemm16_kernel(const _Float16* __restrict__ A, const _Float16* __restrict__ Bt,
                   const float* __restrict__ bias, float* __restrict__ Cf,
                   _Float16* __restrict__ C16, int M, int N, int K,
                   float outScale, int gelu, int outT) {
  __shared__ _Float16 As[2][64 * 40];
  __shared__ _Float16 Bs[2][64 * 40];

  const int tid = threadIdx.x, lane = tid & 31, wave = tid >> 5;
  const int wm = (wave >> 1) * 32, wn = (wave & 1) * 32;
  const int m0 = blockIdx.y * 64, n0 = blockIdx.x * 64;
  const int nk = K / 32;

  v8f acc[2][2];
  acc[0][0] = v8f_zero(); acc[0][1] = v8f_zero();
  acc[1][0] = v8f_zero(); acc[1][1] = v8f_zero();

  const _Float16* Ab = A + (size_t)m0 * K;
  const _Float16* Bb = Bt + (size_t)n0 * K;

  tile_copy64x32(Ab, K, As[0]);          // prologue: tile 0 in flight
  tile_copy64x32(Bb, K, Bs[0]);

  for (int i = 0; i < nk; ++i) {
    const int cur = i & 1;
    if (i + 1 < nk) {                    // overlap: issue tile i+1
      tile_copy64x32(Ab + (i + 1) * 32, K, As[cur ^ 1]);
      tile_copy64x32(Bb + (i + 1) * 32, K, Bs[cur ^ 1]);
      WAIT_ASYNC(8);                     // tile i (oldest 8 ops) complete
    } else {
      WAIT_ASYNC(0);
    }
    __syncthreads();

    v16h a0 = lds_fragA(As[cur], wm,      0, 40);
    v16h a1 = lds_fragA(As[cur], wm + 16, 0, 40);
    v16h b0 = lds_fragB(Bs[cur], wn,      0, 40);
    v16h b1 = lds_fragB(Bs[cur], wn + 16, 0, 40);
    acc[0][0] = wmma_f16(a0, b0, acc[0][0]);
    acc[0][1] = wmma_f16(a0, b1, acc[0][1]);
    acc[1][0] = wmma_f16(a1, b0, acc[1][0]);
    acc[1][1] = wmma_f16(a1, b1, acc[1][1]);
    __syncthreads();                     // tile i consumed; buffer reusable
  }

  const int rhi = (lane >> 4) << 3;
#pragma unroll
  for (int ti = 0; ti < 2; ++ti)
#pragma unroll
    for (int tj = 0; tj < 2; ++tj)
#pragma unroll
      for (int jj = 0; jj < 8; ++jj) {
        int gm = m0 + wm + ti * 16 + jj + rhi;
        int gn = n0 + wn + tj * 16 + (lane & 15);
        float v = (acc[ti][tj][jj] + bias[gn]) * outScale;
        if (gelu) v = 0.5f * v * (1.0f + erff(v * 0.70710678118654752f));
        if (Cf) Cf[(size_t)gm * N + gn] = v;
        if (C16) {
          if (outT) {  // V: [B,H,DK,SEQ]
            int b = gm >> 11, s = gm & 2047, h = gn >> 7, kk = gn & 127;
            C16[(((size_t)b * NH + h) * DK + kk) * SEQ + s] = (_Float16)v;
          } else {
            C16[(size_t)gm * N + gn] = (_Float16)v;
          }
        }
      }
}

// ---------------- Fused flash attention (f16 in/out) -----------------------
// grid (S/64, B*H); block 128 (4 waves). Q pre-scaled by 1/sqrt(dk).
// Q,K: [B*S][DM] head-concat f16; vT: [B,H,DK,SEQ] f16; out z16 [B*S][DM].

__global__ __launch_bounds__(128)
void flash_attn_kernel(const _Float16* __restrict__ Q, const _Float16* __restrict__ Kp,
                       const _Float16* __restrict__ vT, const int* __restrict__ mask,
                       _Float16* __restrict__ Z) {
  __shared__ _Float16 Qs[64 * 136];
  __shared__ _Float16 Ks[32 * 136];
  __shared__ _Float16 Vs[128 * 40];     // [n=depth][k=kv]
  __shared__ _Float16 Ps[64 * 40];
  __shared__ float    Ss[64 * 36];
  __shared__ int      Ms[64 * 32];
  __shared__ float    mrow[64], lrow[64], arow[64];

  const int tid = threadIdx.x, lane = tid & 31, wave = tid >> 5;
  const int rw0 = wave * 16, rhi = (lane >> 4) << 3;
  const int qt = blockIdx.x, b = blockIdx.y >> 3, h = blockIdx.y & 7;
  const int q0 = qt * 64;
  const size_t tokb = (size_t)b * SEQ;
  const int hoff = h * DK;
  const _Float16* vTb = vT + ((size_t)(b * NH + h)) * DK * SEQ;

  // Q tile (64x128): raw async copy, already scaled.
#pragma unroll
  for (int it = 0; it < 8; ++it) {
    int idx = tid + it * 128;
    int r = idx >> 4, seg = (idx & 15) << 3;           // 16B segments
    async_b128(Q + (tokb + q0 + r) * DM + hoff + seg, &Qs[r * 136 + seg]);
  }
  if (tid < 64) { mrow[tid] = -1.0e30f; lrow[tid] = 0.0f; }

  v8f o[8];
#pragma unroll
  for (int t = 0; t < 8; ++t) o[t] = v8f_zero();

  WAIT_ASYNC(0);
  __syncthreads();

  for (int kt = 0; kt < SEQ / 32; ++kt) {
    // K tile 32x128, V tile 128x32 (rows of vT), mask tile 64x32 ints.
#pragma unroll
    for (int it = 0; it < 4; ++it) {
      int idx = tid + it * 128;
      int r = idx >> 4, seg = (idx & 15) << 3;
      async_b128(Kp + (tokb + kt * 32 + r) * DM + hoff + seg, &Ks[r * 136 + seg]);
    }
#pragma unroll
    for (int it = 0; it < 4; ++it) {
      int idx = tid + it * 128;
      int r = idx >> 2, seg = (idx & 3) << 3;
      async_b128(vTb + (size_t)r * SEQ + kt * 32 + seg, &Vs[r * 40 + seg]);
    }
#pragma unroll
    for (int it = 0; it < 4; ++it) {
      int idx = tid + it * 128;
      int r = idx >> 3, seg = (idx & 7) << 2;          // 4 ints = 16B
      async_b128(mask + ((size_t)b * SEQ + q0 + r) * SEQ + kt * 32 + seg,
                 &Ms[r * 32 + seg]);
    }
    WAIT_ASYNC(0);
    __syncthreads();

    // S band = Qband (16x128) x K^T: 2 n-tiles, 4 K-steps.
    v8f st[2]; st[0] = v8f_zero(); st[1] = v8f_zero();
#pragma unroll
    for (int k0 = 0; k0 < 128; k0 += 32) {
      v16h aq = lds_fragA(Qs, rw0, k0, 136);
      st[0] = wmma_f16(aq, lds_fragB(Ks, 0,  k0, 136), st[0]);
      st[1] = wmma_f16(aq, lds_fragB(Ks, 16, k0, 136), st[1]);
    }
#pragma unroll
    for (int t = 0; t < 2; ++t)
#pragma unroll
      for (int jj = 0; jj < 8; ++jj) {
        int mb = rw0 + jj + rhi;
        int n  = t * 16 + (lane & 15);
        Ss[mb * 36 + n] = (Ms[mb * 32 + n] == 0) ? NEGV : st[t][jj];
      }
    __syncthreads();

    // Online softmax, one thread per query row.
    if (tid < 64) {
      float mp = mrow[tid], mx = mp;
#pragma unroll
      for (int j = 0; j < 32; ++j) mx = fmaxf(mx, Ss[tid * 36 + j]);
      float al = expf(mp - mx), sum = 0.0f;
#pragma unroll
      for (int j = 0; j < 32; ++j) {
        float p = expf(Ss[tid * 36 + j] - mx);
        Ps[tid * 40 + j] = (_Float16)p;
        sum += p;
      }
      lrow[tid] = lrow[tid] * al + sum;
      mrow[tid] = mx; arow[tid] = al;
    }
    __syncthreads();

    // Rescale O, accumulate P (16x32) x V (32x128).
    float af[8];
#pragma unroll
    for (int jj = 0; jj < 8; ++jj) af[jj] = arow[rw0 + jj + rhi];
    v16h ap = lds_fragA(Ps, rw0, 0, 40);
#pragma unroll
    for (int t = 0; t < 8; ++t) {
#pragma unroll
      for (int jj = 0; jj < 8; ++jj) o[t][jj] *= af[jj];
      o[t] = wmma_f16(ap, lds_fragB(Vs, t * 16, 0, 40), o[t]);
    }
    __syncthreads();                     // Ks/Vs/Ms/Ps free for next kt
  }

  float lf[8];
#pragma unroll
  for (int jj = 0; jj < 8; ++jj) lf[jj] = lrow[rw0 + jj + rhi];
#pragma unroll
  for (int t = 0; t < 8; ++t)
#pragma unroll
    for (int jj = 0; jj < 8; ++jj) {
      int gm = q0 + rw0 + jj + rhi;
      int col = hoff + t * 16 + (lane & 15);
      Z[(tokb + gm) * DM + col] = (_Float16)(o[t][jj] / lf[jj]);
    }
}

// ---------------- Residual add + LayerNorm (f32 + f16 outputs) -------------

__global__ __launch_bounds__(256)
void addln_kernel(const float* __restrict__ x, const float* __restrict__ y,
                  const float* __restrict__ g, const float* __restrict__ bta,
                  float* __restrict__ outF, _Float16* __restrict__ out16) {
  __shared__ float red[256];
  __shared__ float s_mean, s_rstd;
  const int tid = threadIdx.x;
  const size_t base = (size_t)blockIdx.x * DM;

  float loc[4]; float s = 0.0f;
#pragma unroll
  for (int j = 0; j < 4; ++j) {
    int c = tid + j * 256;
    loc[j] = x[base + c] + y[base + c];
    s += loc[j];
  }
  red[tid] = s; __syncthreads();
  for (int off = 128; off > 0; off >>= 1) {
    if (tid < off) red[tid] += red[tid + off];
    __syncthreads();
  }
  if (tid == 0) s_mean = red[0] * (1.0f / DM);
  __syncthreads();
  float m = s_mean; s = 0.0f;
#pragma unroll
  for (int j = 0; j < 4; ++j) { float d = loc[j] - m; s += d * d; }
  red[tid] = s; __syncthreads();
  for (int off = 128; off > 0; off >>= 1) {
    if (tid < off) red[tid] += red[tid + off];
    __syncthreads();
  }
  if (tid == 0) s_rstd = rsqrtf(red[0] * (1.0f / DM) + 1e-5f);
  __syncthreads();
  float r = s_rstd;
#pragma unroll
  for (int j = 0; j < 4; ++j) {
    int c = tid + j * 256;
    float v = (loc[j] - m) * r * g[c] + bta[c];
    outF[base + c] = v;
    out16[base + c] = (_Float16)v;
  }
}

// ---------------- One-time weight transpose + f32->f16 ---------------------
// Wt[n][k] = W(k,n); headStride>0: W(k,n)=w[(n>>7)*hs + k*128 + (n&127)].

__global__ __launch_bounds__(256)
void wconv_kernel(const float* __restrict__ W, _Float16* __restrict__ Wt,
                  int K, int N, long long headStride) {
  __shared__ float t[32][33];
  const int nb = blockIdx.x * 32, kb = blockIdx.y * 32;
  const int c = threadIdx.x & 31, r0 = threadIdx.x >> 5;
#pragma unroll
  for (int rr = 0; rr < 32; rr += 8) {
    int k = kb + r0 + rr, n = nb + c;
    size_t off = headStride
        ? ((size_t)(n >> 7) * headStride + (size_t)k * 128 + (n & 127))
        : ((size_t)k * N + n);
    t[r0 + rr][c] = W[off];
  }
  __syncthreads();
#pragma unroll
  for (int rr = 0; rr < 32; rr += 8) {
    int n = nb + r0 + rr, k = kb + c;
    Wt[(size_t)n * K + k] = (_Float16)t[c][r0 + rr];
  }
}

__global__ __launch_bounds__(256)
void cvt16_kernel(const float* __restrict__ x, _Float16* __restrict__ y) {
  int i = blockIdx.x * 256 + threadIdx.x;
  float4 v = ((const float4*)x)[i];
  v4h o; o[0] = (_Float16)v.x; o[1] = (_Float16)v.y;
         o[2] = (_Float16)v.z; o[3] = (_Float16)v.w;
  *(v4h*)(y + 4 * (size_t)i) = o;
}

// ---------------- Host orchestration ---------------------------------------

static void launch_gemm(const _Float16* A, const _Float16* Bt, const float* bias,
                        float* Cf, _Float16* C16, float scale, int gelu, int outT,
                        hipStream_t stream) {
  dim3 grid(DM / 64, MTOK / 64), block(128);
  gemm16_kernel<<<grid, block, 0, stream>>>(A, Bt, bias, Cf, C16,
                                            MTOK, DM, DM, scale, gelu, outT);
}

extern "C" void kernel_launch(void* const* d_in, const int* in_sizes, int n_in,
                              void* d_out, int out_size, void* d_ws, size_t ws_size,
                              hipStream_t stream) {
  const float* x_in  = (const float*)d_in[0];
  const int*   amask = (const int*)d_in[1];
  const float* wq = (const float*)d_in[2];
  const float* bq = (const float*)d_in[3];
  const float* wk = (const float*)d_in[4];
  const float* bk = (const float*)d_in[5];
  const float* wv = (const float*)d_in[6];
  const float* bv = (const float*)d_in[7];
  const float* wo = (const float*)d_in[8];
  const float* bo = (const float*)d_in[9];
  const float* w1 = (const float*)d_in[10];
  const float* b1 = (const float*)d_in[11];
  const float* w2 = (const float*)d_in[12];
  const float* b2 = (const float*)d_in[13];
  const float* ln1g = (const float*)d_in[14];
  const float* ln1b = (const float*)d_in[15];
  const float* ln2g = (const float*)d_in[16];
  const float* ln2b = (const float*)d_in[17];
  (void)in_sizes; (void)n_in; (void)out_size; (void)ws_size;

  const size_t WMAT = (size_t)DM * DM;        // 1M elements
  const size_t ABUF = (size_t)MTOK * DM;      // 8M elements
  char* p = (char*)d_ws;
  _Float16* wT = (_Float16*)p;                p += 6 * NL * WMAT * 2;   // 48MB
  _Float16* xA16 = (_Float16*)p;              p += ABUF * 2;
  _Float16* xB16 = (_Float16*)p;              p += ABUF * 2;
  _Float16* q16  = (_Float16*)p;              p += ABUF * 2;
  _Float16* k16  = (_Float16*)p;              p += ABUF * 2;
  _Float16* vTb  = (_Float16*)p;              p += ABUF * 2;
  _Float16* z16  = (_Float16*)p;              p += ABUF * 2;
  _Float16* h16  = (_Float16*)p;              p += ABUF * 2;
  float*    t0f  = (float*)p;                 p += ABUF * 4;
  float*    xAf  = (float*)p;                 p += ABUF * 4;
  float*    xBf  = (float*)p;                 p += ABUF * 4;

  const size_t wQkv = (size_t)NH * DM * DK;   // per-layer qkv weight elems
  const long long HS = (long long)DM * DK;    // head stride
  auto WT = [&](int l, int j) { return wT + ((size_t)l * 6 + j) * WMAT; };

  // One-time weight transpose/convert + input f16 copy.
  {
    dim3 cg(DM / 32, DM / 32), cb(256);
    for (int l = 0; l < NL; ++l) {
      wconv_kernel<<<cg, cb, 0, stream>>>(wq + l * wQkv, WT(l, 0), DM, DM, HS);
      wconv_kernel<<<cg, cb, 0, stream>>>(wk + l * wQkv, WT(l, 1), DM, DM, HS);
      wconv_kernel<<<cg, cb, 0, stream>>>(wv + l * wQkv, WT(l, 2), DM, DM, HS);
      wconv_kernel<<<cg, cb, 0, stream>>>(wo + l * WMAT, WT(l, 3), DM, DM, 0);
      wconv_kernel<<<cg, cb, 0, stream>>>(w1 + l * WMAT, WT(l, 4), DM, DM, 0);
      wconv_kernel<<<cg, cb, 0, stream>>>(w2 + l * WMAT, WT(l, 5), DM, DM, 0);
    }
    cvt16_kernel<<<dim3(ABUF / 4 / 256), cb, 0, stream>>>(x_in, xB16);
  }

  const float qscale = 0.08838834764831845f;  // 1/sqrt(128)
  const float* curF = x_in;
  const _Float16* cur16 = xB16;

  for (int i = 0; i < NL; ++i) {
    launch_gemm(cur16, WT(i, 0), bq + i * DM, nullptr, q16, qscale, 0, 0, stream);
    launch_gemm(cur16, WT(i, 1), bk + i * DM, nullptr, k16, 1.0f,   0, 0, stream);
    launch_gemm(cur16, WT(i, 2), bv + i * DM, nullptr, vTb, 1.0f,   0, 1, stream);

    {
      dim3 grid(SEQ / 64, NB * NH), block(128);
      flash_attn_kernel<<<grid, block, 0, stream>>>(q16, k16, vTb, amask, z16);
    }

    launch_gemm(z16, WT(i, 3), bo + i * DM, t0f, nullptr, 1.0f, 0, 0, stream);
    addln_kernel<<<dim3(MTOK), dim3(256), 0, stream>>>(
        curF, t0f, ln1g + i * DM, ln1b + i * DM, xBf, xB16);

    launch_gemm(xB16, WT(i, 4), b1 + i * DM, nullptr, h16, 1.0f, 1, 0, stream);
    launch_gemm(h16,  WT(i, 5), b2 + i * DM, t0f, nullptr, 1.0f, 0, 0, stream);

    float* dstF = (i == NL - 1) ? (float*)d_out : xAf;
    addln_kernel<<<dim3(MTOK), dim3(256), 0, stream>>>(
        xBf, t0f, ln2g + i * DM, ln2b + i * DM, dstF, xA16);
    curF = xAf; cur16 = xA16;
  }
}
